// MgSmmSModel_16767552323947
// MI455X (gfx1250) — compile-verified, matrix operand order checked by reading, not run on hardware
//
#include <hip/hip_runtime.h>

typedef __attribute__((ext_vector_type(16))) __bf16 v16bf;
typedef __attribute__((ext_vector_type(8)))  __bf16 v8bf;
typedef __attribute__((ext_vector_type(8)))  float  v8f;
typedef __attribute__((__vector_size__(4 * sizeof(int)))) int i32x4;

#define B_     64
#define S_     512
#define H_     4096
#define G_     2048
#define OUT_   64
#define NWG    32
#define TPB    256
#define KCHUNK 128
#define NCHUNK (H_ / KCHUNK)        // 32
#define KT     (KCHUNK / 32)        // 4 wmma-K iterations per chunk
#define PITCH  (KCHUNK + 8)         // 136 bf16 = 272 B row pitch (16B aligned, bank-skewed)

#ifndef __has_builtin
#define __has_builtin(x) 0
#endif
#if __has_builtin(__builtin_amdgcn_global_load_async_to_lds_b128)
#define HAVE_ASYNC_LDS 1
#else
#define HAVE_ASYNC_LDS 0
#endif

#define AS3 __attribute__((address_space(3)))

// ---------------- workspace layout (bytes) ----------------
static constexpr size_t OFF_WA  = 0;                                  // H*H bf16
static constexpr size_t OFF_WC  = OFF_WA + (size_t)H_ * H_ * 2;       // OUT*H bf16
static constexpr size_t OFF_H0  = OFF_WC + (size_t)OUT_ * H_ * 2;     // B*H bf16
static constexpr size_t OFF_H1  = OFF_H0 + (size_t)B_ * H_ * 2;       // B*H bf16
static constexpr size_t OFF_CB  = OFF_H1 + (size_t)B_ * H_ * 2;       // H f32
static constexpr size_t OFF_BAR = OFF_CB + (size_t)H_ * 4;            // 2 u32

static __device__ inline v16bf cat16(v8bf lo, v8bf hi) {
  return __builtin_shufflevector(lo, hi, 0,1,2,3,4,5,6,7,8,9,10,11,12,13,14,15);
}

// copy 16 B global -> LDS (async-to-LDS when the toolchain exposes it)
static __device__ inline void cp16_g2l(const __bf16* gsrc, __bf16* ldst) {
#if HAVE_ASYNC_LDS
  __builtin_amdgcn_global_load_async_to_lds_b128(
      (i32x4*)(uintptr_t)gsrc,
      (AS3 i32x4*)(uint32_t)(uintptr_t)ldst, 0, 0);
#else
  *reinterpret_cast<uint4*>(ldst) = *reinterpret_cast<const uint4*>(gsrc);
#endif
}

static __device__ inline void wait_async0() {
#if HAVE_ASYNC_LDS
# if __has_builtin(__builtin_amdgcn_s_wait_asynccnt)
  __builtin_amdgcn_s_wait_asynccnt(0);
# else
  asm volatile("s_wait_asynccnt 0" ::: "memory");
# endif
#endif
}

// ---------------- prologue kernels ----------------
__global__ void k_cvt4(const float* __restrict__ src, __bf16* __restrict__ dst, int n4) {
  int i = blockIdx.x * blockDim.x + threadIdx.x;
  if (i < n4) {
    float4 v = reinterpret_cast<const float4*>(src)[i];
    __bf16* d = dst + (size_t)i * 4;
    d[0] = (__bf16)v.x; d[1] = (__bf16)v.y; d[2] = (__bf16)v.z; d[3] = (__bf16)v.w;
  }
}

__global__ void k_prep(const float* __restrict__ bA, const float* __restrict__ bB,
                       const float* __restrict__ Wbh, float* __restrict__ cbias,
                       unsigned* __restrict__ bar) {
  int i = blockIdx.x * blockDim.x + threadIdx.x;
  if (i < H_) cbias[i] = bA[i] + bB[i] + Wbh[i];
  if (i == 0) { bar[0] = 0u; bar[1] = 0u; }
}

__global__ void k_zero32(unsigned* __restrict__ p, int n) {
  int i = blockIdx.x * blockDim.x + threadIdx.x;
  if (i < n) p[i] = 0u;
}

// ---------------- device-wide barrier ----------------
static __device__ inline void gridBarrier(unsigned* cnt, unsigned* gen) {
  __syncthreads();
  if (threadIdx.x == 0) {
    unsigned g0 = __hip_atomic_load(gen, __ATOMIC_RELAXED, __HIP_MEMORY_SCOPE_AGENT);
    unsigned a  = __hip_atomic_fetch_add(cnt, 1u, __ATOMIC_ACQ_REL, __HIP_MEMORY_SCOPE_AGENT);
    if (a == NWG - 1) {
      __hip_atomic_store(cnt, 0u, __ATOMIC_RELAXED, __HIP_MEMORY_SCOPE_AGENT);
      __hip_atomic_fetch_add(gen, 1u, __ATOMIC_ACQ_REL, __HIP_MEMORY_SCOPE_AGENT);
    } else {
      while (__hip_atomic_load(gen, __ATOMIC_ACQUIRE, __HIP_MEMORY_SCOPE_AGENT) == g0)
        __builtin_amdgcn_s_sleep(1);
    }
  }
  __syncthreads();
}

// ---------------- persistent scan + epilogue ----------------
__global__ __launch_bounds__(TPB, 1)
void k_scan(const float* __restrict__ x,        // (B,S,1)
            const float* __restrict__ WB,       // (H,1)
            const float* __restrict__ cbias,    // (H)
            const __bf16* __restrict__ WAbf,    // (H,H) row-major bf16
            __bf16* __restrict__ h0,            // (B,H) bf16 ping (zero-initialized)
            __bf16* __restrict__ h1,            // (B,H) bf16 pong
            unsigned* __restrict__ bar,
            const __bf16* __restrict__ WCbf,    // (OUT,H) bf16
            const float* __restrict__ bC, const float* __restrict__ bD,
            const float* __restrict__ bJ, const float* __restrict__ WD,
            const float* __restrict__ WJ,       // (OUT,G)
            float* __restrict__ out)            // (B,OUT)
{
  __shared__ __bf16 ldsH[2][B_][PITCH];
  __shared__ float  ldsX[B_];

  const int tid  = threadIdx.x;
  const int wg   = blockIdx.x;
  const int lane = tid & 31;
  const int w    = tid >> 5;          // wave in workgroup, 0..7
  const int lm   = lane & 15;
  const int g    = lane >> 4;         // K-group select per ISA 16-bit fragment layout

  const int nglob = (wg * 8 + w) * 16 + lm;       // this lane's output column (fixed)
  const __bf16* browBase = WAbf + (size_t)nglob * H_ + g * 8;
  const float wb = WB[nglob];
  const float cb = cbias[nglob];

  // staging: thread copies row sr, segment sq (32 bf16 = 64 B = 4 x b128)
  const int sr = tid >> 2;            // 0..63
  const int sq = tid & 3;             // 0..3

  for (int t = 0; t < S_; ++t) {
    const __bf16* hc = (t & 1) ? h1 : h0;
    __bf16*       hn = (t & 1) ? h0 : h1;

    // ---- stage chunk 0 into buffer 0 ----
    {
      const __bf16* gsrc = hc + (size_t)sr * H_ + sq * 32;
      __bf16* ldst = &ldsH[0][sr][sq * 32];
#pragma unroll
      for (int i = 0; i < 4; ++i) cp16_g2l(gsrc + i * 8, ldst + i * 8);
    }
    if (tid < B_) ldsX[tid] = x[(size_t)tid * S_ + t];
    wait_async0();
    __syncthreads();

    // ---- C init = drive[t] = x[b,t]*W_B[n] + (b_A+b_B+W_bh)[n] ----
    v8f acc[4];
#pragma unroll
    for (int mt = 0; mt < 4; ++mt)
#pragma unroll
      for (int v = 0; v < 8; ++v)
        acc[mt][v] = ldsX[mt * 16 + g * 8 + v] * wb + cb;

    for (int c = 0; c < NCHUNK; ++c) {
      const int cur = c & 1;

      // prefetch next chunk into the other buffer while we compute
      if (c + 1 < NCHUNK) {
        const __bf16* gsrc = hc + (size_t)sr * H_ + (c + 1) * KCHUNK + sq * 32;
        __bf16* ldst = &ldsH[cur ^ 1][sr][sq * 32];
#pragma unroll
        for (int i = 0; i < 4; ++i) cp16_g2l(gsrc + i * 8, ldst + i * 8);
      }

      // ---- compute on buffer `cur`, software-pipelined A-fragment loads ----
      const __bf16* bbase = browBase + c * KCHUNK;
      v16bf af[4];
#pragma unroll
      for (int mt = 0; mt < 4; ++mt) {
        const __bf16* ap = &ldsH[cur][mt * 16 + lm][g * 8];
        af[mt] = cat16(*(const v8bf*)ap, *(const v8bf*)(ap + 16));
      }
#pragma unroll
      for (int kt = 0; kt < KT; ++kt) {
        const int kb = kt * 32;
        v16bf bf = cat16(*(const v8bf*)(bbase + kb),
                         *(const v8bf*)(bbase + kb + 16));
        v16bf afn[4];
        if (kt + 1 < KT) {
#pragma unroll
          for (int mt = 0; mt < 4; ++mt) {
            const __bf16* ap = &ldsH[cur][mt * 16 + lm][kb + 32 + g * 8];
            afn[mt] = cat16(*(const v8bf*)ap, *(const v8bf*)(ap + 16));
          }
        }
#pragma unroll
        for (int mt = 0; mt < 4; ++mt)
          acc[mt] = __builtin_amdgcn_wmma_f32_16x16x32_bf16(
              false, af[mt], false, bf, (short)0, acc[mt], false, false);
        if (kt + 1 < KT) {
#pragma unroll
          for (int mt = 0; mt < 4; ++mt) af[mt] = afn[mt];
        }
      }

      if (c + 1 < NCHUNK) {
        wait_async0();
        __syncthreads();
      }
    }

    // ---- write h_next per C/D fragment layout: lanes0-15 M=v, lanes16-31 M=v+8 ----
#pragma unroll
    for (int mt = 0; mt < 4; ++mt)
#pragma unroll
      for (int v = 0; v < 8; ++v) {
        const int m = mt * 16 + g * 8 + v;
        hn[(size_t)m * H_ + nglob] = (__bf16)acc[mt][v];
      }

    __threadfence();
    gridBarrier(bar, bar + 1);
  }

  // -------- epilogue: out = h_T@W_C^T + b_C + x_last@W_D^T + b_D + rowsum(W_J) + b_J --------
  if (wg != 0) return;
  const __bf16* hfin = h0;  // after 512 steps (even), h lives in buffer 0

  for (int tt = w * 2; tt < w * 2 + 2; ++tt) {
    const int mt = tt >> 2, ot = tt & 3;
    const int o  = ot * 16 + lm;

    // g_T is all-ones => contribution is rowsum of W_J
    float js = 0.f;
    const float4* jr = reinterpret_cast<const float4*>(WJ + (size_t)o * G_);
    for (int i = 0; i < G_ / 4; ++i) { float4 q = jr[i]; js += q.x + q.y + q.z + q.w; }

    v8f acc = {0.f, 0.f, 0.f, 0.f, 0.f, 0.f, 0.f, 0.f};
    const __bf16* bro = WCbf + (size_t)o * H_ + g * 8;
    const __bf16* aro = hfin + (size_t)(mt * 16 + lm) * H_ + g * 8;
    for (int kb = 0; kb < H_; kb += 32) {
      v16bf bf = cat16(*(const v8bf*)(bro + kb), *(const v8bf*)(bro + kb + 16));
      v16bf af = cat16(*(const v8bf*)(aro + kb), *(const v8bf*)(aro + kb + 16));
      acc = __builtin_amdgcn_wmma_f32_16x16x32_bf16(
          false, af, false, bf, (short)0, acc, false, false);
    }

    const float base = bC[o] + bD[o] + bJ[o] + js;
    const float wd   = WD[o];
#pragma unroll
    for (int v = 0; v < 8; ++v) {
      const int m = mt * 16 + g * 8 + v;
      out[(size_t)m * OUT_ + o] = acc[v] + base + x[(size_t)m * S_ + (S_ - 1)] * wd;
    }
  }
}

// ---------------- host launcher ----------------
extern "C" void kernel_launch(void* const* d_in, const int* in_sizes, int n_in,
                              void* d_out, int out_size, void* d_ws, size_t ws_size,
                              hipStream_t stream) {
  (void)in_sizes; (void)n_in; (void)out_size; (void)ws_size;
  const float* x   = (const float*)d_in[0];
  const float* WA  = (const float*)d_in[1];
  const float* bA  = (const float*)d_in[2];
  const float* WB  = (const float*)d_in[3];
  const float* bB  = (const float*)d_in[4];
  const float* Wbh = (const float*)d_in[5];
  const float* WC  = (const float*)d_in[11];
  const float* bC  = (const float*)d_in[12];
  const float* WD  = (const float*)d_in[13];
  const float* bD  = (const float*)d_in[14];
  const float* WJ  = (const float*)d_in[15];
  const float* bJ  = (const float*)d_in[16];

  char* ws = (char*)d_ws;
  __bf16*   WAbf  = (__bf16*)(ws + OFF_WA);
  __bf16*   WCbf  = (__bf16*)(ws + OFF_WC);
  __bf16*   h0    = (__bf16*)(ws + OFF_H0);
  __bf16*   h1    = (__bf16*)(ws + OFF_H1);
  float*    cbias = (float*)(ws + OFF_CB);
  unsigned* bar   = (unsigned*)(ws + OFF_BAR);

  // convert weights to bf16 (one-time; L2-resident thereafter)
  k_cvt4<<<(H_ * H_ / 4 + TPB - 1) / TPB, TPB, 0, stream>>>(WA, WAbf, H_ * H_ / 4);
  k_cvt4<<<(OUT_ * H_ / 4 + TPB - 1) / TPB, TPB, 0, stream>>>(WC, WCbf, OUT_ * H_ / 4);
  // fused bias + barrier reset
  k_prep<<<(H_ + TPB - 1) / TPB, TPB, 0, stream>>>(bA, bB, Wbh, cbias, bar);
  // h0 = 0 (B*H bf16 = B*H/2 u32)
  k_zero32<<<(B_ * H_ / 2 + TPB - 1) / TPB, TPB, 0, stream>>>((unsigned*)h0, B_ * H_ / 2);

  // persistent cooperative scan (512 steps) + WMMA epilogue
  k_scan<<<NWG, TPB, 0, stream>>>(x, WB, cbias, WAbf, h0, h1, bar,
                                  WCbf, bC, bD, bJ, WD, WJ, (float*)d_out);
}
